// TemporalContrastiveLoss_5909875000351
// MI455X (gfx1250) — matrix operand order repaired and checked by reading, not compiled, not run on hardware
//
#include <hip/hip_runtime.h>
#include <hip/hip_bf16.h>
#include <math.h>

// Problem constants from the reference (D fixed by the model; M derived from in_sizes).
#define DIM         128
#define TILE        16
#define NEGS        5
#define INV_TEMP    2.0f      // 1 / TEMPERATURE, TEMPERATURE = 0.5
#define EPS_NORM    1e-12f

typedef __attribute__((ext_vector_type(16))) _Float16 v16h;
typedef __attribute__((ext_vector_type(8)))  _Float16 v8h;
typedef __attribute__((ext_vector_type(4)))  _Float16 v4h;
typedef __attribute__((ext_vector_type(8)))  float    v8f;

__device__ __forceinline__ float vget8(v8f v, int e) {
    switch (e) {
        case 0: return v[0]; case 1: return v[1]; case 2: return v[2]; case 3: return v[3];
        case 4: return v[4]; case 5: return v[5]; case 6: return v[6]; default: return v[7];
    }
}

// Shared tail: diagonal extraction + logsumexp + wave reduce + per-tile partial.
__device__ __forceinline__ void tcl_finish(const v8f acc[6], int lane, int m0, int MA,
                                           float* __restrict__ partials, int wave)
{
    // Diagonal (i,i) of 16x16 f32 D: i<8 -> lane i elem i; i>=8 -> lane i+16 elem i-8.
    float contrib = 0.0f;
    int ai = -1, elem = 0;
    if (lane < 8)        { ai = lane;      elem = lane;      }
    else if (lane >= 24) { ai = lane - 16; elem = lane - 24; }

    if (ai >= 0) {
        const int ga = m0 + ai;
        if (ga < MA) {
            float l[6];
            #pragma unroll
            for (int t = 0; t < 6; ++t) l[t] = INV_TEMP * vget8(acc[t], elem);
            float mx = l[0];
            #pragma unroll
            for (int t = 1; t < 6; ++t) mx = fmaxf(mx, l[t]);
            float s = 0.0f;
            #pragma unroll
            for (int t = 0; t < 6; ++t) s += expf(l[t] - mx);
            contrib = mx + logf(s) - l[0];       // logsumexp - pos_logit
        }
    }
    #pragma unroll
    for (int off = 16; off > 0; off >>= 1)
        contrib += __shfl_xor(contrib, off, 32);
    if (lane == 0)
        partials[wave] = contrib;
}

// ---------------------------------------------------------------------------
// Path A, pass 1: normalize rows AND write a normalized-f16 mirror (64 MB).
// One wave32 per row; butterfly reduce gives the norm to every lane.
// ---------------------------------------------------------------------------
__global__ __launch_bounds__(256)
void tcl_norm_f16_kernel(const float* __restrict__ emb,
                         _Float16* __restrict__ embh, int M)
{
    const int lane = threadIdx.x & 31;
    const int row  = blockIdx.x * (blockDim.x >> 5) + (threadIdx.x >> 5);
    if (row >= M) return;

    const unsigned rowOffF = (unsigned)row * (DIM * 4u);     // f32 bytes
    const float4 v = *(const float4*)((const char*)emb + rowOffF + lane * 16u);
    float ss = v.x * v.x + v.y * v.y + v.z * v.z + v.w * v.w;
    #pragma unroll
    for (int off = 16; off > 0; off >>= 1)
        ss += __shfl_xor(ss, off, 32);
    const float inv = 1.0f / fmaxf(sqrtf(ss), EPS_NORM);

    v4h h;
    h[0] = (_Float16)(v.x * inv); h[1] = (_Float16)(v.y * inv);
    h[2] = (_Float16)(v.z * inv); h[3] = (_Float16)(v.w * inv);
    const unsigned rowOffH = (unsigned)row * (DIM * 2u);     // f16 bytes
    *(v4h*)((char*)embh + rowOffH + lane * 8u) = h;          // 8-byte store
}

// ---------------------------------------------------------------------------
// Path A, pass 2: WMMA tiles straight from normalized f16 (no per-element math).
// 32-bit byte offsets from a uniform base -> saddr-form global loads; the chunk
// constant folds into the 24-bit instruction offset.
// ---------------------------------------------------------------------------
__global__ __launch_bounds__(256)
void tcl_wmma_f16_kernel(const _Float16* __restrict__ embh,
                         const int*      __restrict__ negIdx,
                         float*          __restrict__ partials,
                         int M, int MA, int numTiles)
{
    const int lane = threadIdx.x & 31;
    const int wave = blockIdx.x * (blockDim.x >> 5) + (threadIdx.x >> 5);
    if (wave >= numTiles) return;            // uniform whole-wave exit: EXEC stays full

    const int m0 = wave * TILE;
    const int hw = lane >> 4;
    const int lm = lane & 15;
    const char* base = (const char*)embh;

    int rA = m0 + lm; if (rA > M - 1) rA = M - 1;
    const unsigned offA = (unsigned)rA * (DIM * 2u) + (unsigned)(hw * 32);

    int aC = m0 + lm; if (aC > MA - 1) aC = MA - 1;
    unsigned offC[6];
    {
        int cand[6];
        cand[0] = aC + 1;
        #pragma unroll
        for (int k = 0; k < NEGS; ++k)
            cand[1 + k] = negIdx[(size_t)aC * NEGS + k];
        #pragma unroll
        for (int t = 0; t < 6; ++t)
            offC[t] = (unsigned)cand[t] * (DIM * 2u) + (unsigned)(hw * 32);
    }

    v8f acc[6];
    #pragma unroll
    for (int t = 0; t < 6; ++t) { v8f z = {}; acc[t] = z; }

    #pragma unroll
    for (int c = 0; c < 4; ++c) {
        const unsigned kb = (unsigned)(c * 64);  // 32 halves = 64 bytes per chunk

        // A (16-bit 16x32): halves 0..15 = K kb+16*hw+{0..15} (contiguous).
        v8h lo = *(const v8h*)(base + offA + kb);
        v8h hi = *(const v8h*)(base + offA + kb + 16u);
        v16h a = __builtin_shufflevector(lo, hi,
                     0, 1, 2, 3, 4, 5, 6, 7, 8, 9, 10, 11, 12, 13, 14, 15);

        // B (16-bit 32x16): lane = column lm, 16 contiguous halves (32B).
        #pragma unroll
        for (int t = 0; t < 6; ++t) {
            v16h bv = *(const v16h*)(base + offC[t] + kb);
            acc[t] = __builtin_amdgcn_wmma_f32_16x16x32_f16(
                         false, a, false, bv, (short)0, acc[t], false, false);
        }
    }

    tcl_finish(acc, lane, m0, MA, partials, wave);
}

// ---------------------------------------------------------------------------
// Path B (small-ws fallback): inv-norm table + on-the-fly f32->f16 convert.
// ---------------------------------------------------------------------------
__global__ __launch_bounds__(256)
void tcl_norm_kernel(const float* __restrict__ emb, float* __restrict__ invNorm, int M)
{
    const int lane = threadIdx.x & 31;
    const int row  = blockIdx.x * (blockDim.x >> 5) + (threadIdx.x >> 5);
    if (row >= M) return;
    const float4 v = *(const float4*)((const char*)emb +
                         (unsigned)row * (DIM * 4u) + lane * 16u);
    float ss = v.x * v.x + v.y * v.y + v.z * v.z + v.w * v.w;
    #pragma unroll
    for (int off = 16; off > 0; off >>= 1)
        ss += __shfl_xor(ss, off, 32);
    if (lane == 0)
        invNorm[row] = 1.0f / fmaxf(sqrtf(ss), EPS_NORM);
}

__global__ __launch_bounds__(256)
void tcl_wmma_f32_kernel(const float* __restrict__ emb,
                         const int*   __restrict__ negIdx,
                         const float* __restrict__ invNorm,
                         float*       __restrict__ partials,
                         int M, int MA, int numTiles)
{
    const int lane = threadIdx.x & 31;
    const int wave = blockIdx.x * (blockDim.x >> 5) + (threadIdx.x >> 5);
    if (wave >= numTiles) return;

    const int m0 = wave * TILE;
    const int hw = lane >> 4;
    const int lm = lane & 15;
    const char* base = (const char*)emb;

    int rA = m0 + lm; if (rA > M - 1) rA = M - 1;
    const unsigned offA = (unsigned)rA * (DIM * 4u) + (unsigned)(hw * 64);
    const float invA = invNorm[rA];

    int aC = m0 + lm; if (aC > MA - 1) aC = MA - 1;
    int cand[6];
    cand[0] = aC + 1;
    #pragma unroll
    for (int k = 0; k < NEGS; ++k)
        cand[1 + k] = negIdx[(size_t)aC * NEGS + k];
    unsigned offC[6]; float invC[6];
    #pragma unroll
    for (int t = 0; t < 6; ++t) {
        offC[t] = (unsigned)cand[t] * (DIM * 4u) + (unsigned)(hw * 64);
        invC[t] = invNorm[cand[t]];
    }

    v8f acc[6];
    #pragma unroll
    for (int t = 0; t < 6; ++t) { v8f z = {}; acc[t] = z; }

    #pragma unroll
    for (int c = 0; c < 4; ++c) {
        const unsigned kb = (unsigned)(c * 128);    // 32 floats = 128 bytes per chunk
        v16h a;
        {
            float4 r0 = *(const float4*)(base + offA + kb);
            float4 r1 = *(const float4*)(base + offA + kb + 16u);
            float4 r2 = *(const float4*)(base + offA + kb + 32u);
            float4 r3 = *(const float4*)(base + offA + kb + 48u);
            a[0]  = (_Float16)(r0.x * invA); a[1]  = (_Float16)(r0.y * invA);
            a[2]  = (_Float16)(r0.z * invA); a[3]  = (_Float16)(r0.w * invA);
            a[4]  = (_Float16)(r1.x * invA); a[5]  = (_Float16)(r1.y * invA);
            a[6]  = (_Float16)(r1.z * invA); a[7]  = (_Float16)(r1.w * invA);
            a[8]  = (_Float16)(r2.x * invA); a[9]  = (_Float16)(r2.y * invA);
            a[10] = (_Float16)(r2.z * invA); a[11] = (_Float16)(r2.w * invA);
            a[12] = (_Float16)(r3.x * invA); a[13] = (_Float16)(r3.y * invA);
            a[14] = (_Float16)(r3.z * invA); a[15] = (_Float16)(r3.w * invA);
        }
        #pragma unroll
        for (int t = 0; t < 6; ++t) {
            float4 b0 = *(const float4*)(base + offC[t] + kb);
            float4 b1 = *(const float4*)(base + offC[t] + kb + 16u);
            float4 b2 = *(const float4*)(base + offC[t] + kb + 32u);
            float4 b3 = *(const float4*)(base + offC[t] + kb + 48u);
            const float iv = invC[t];
            v16h bv;
            bv[0]  = (_Float16)(b0.x * iv); bv[1]  = (_Float16)(b0.y * iv);
            bv[2]  = (_Float16)(b0.z * iv); bv[3]  = (_Float16)(b0.w * iv);
            bv[4]  = (_Float16)(b1.x * iv); bv[5]  = (_Float16)(b1.y * iv);
            bv[6]  = (_Float16)(b1.z * iv); bv[7]  = (_Float16)(b1.w * iv);
            bv[8]  = (_Float16)(b2.x * iv); bv[9]  = (_Float16)(b2.y * iv);
            bv[10] = (_Float16)(b2.z * iv); bv[11] = (_Float16)(b2.w * iv);
            bv[12] = (_Float16)(b3.x * iv); bv[13] = (_Float16)(b3.y * iv);
            bv[14] = (_Float16)(b3.z * iv); bv[15] = (_Float16)(b3.w * iv);
            acc[t] = __builtin_amdgcn_wmma_f32_16x16x32_f16(
                         false, a, false, bv, (short)0, acc[t], false, false);
        }
    }

    tcl_finish(acc, lane, m0, MA, partials, wave);
}

// ---------------------------------------------------------------------------
// Final deterministic single-block reduction.
// ---------------------------------------------------------------------------
__global__ __launch_bounds__(256)
void tcl_reduce_kernel(const float* __restrict__ partials, int n,
                       float* __restrict__ out, float invMA)
{
    __shared__ float sm[256];
    const int tid = threadIdx.x;
    float v = 0.0f;
    for (int i = tid; i < n; i += 256) v += partials[i];
    sm[tid] = v;
    __syncthreads();
    #pragma unroll
    for (int off = 128; off > 0; off >>= 1) {
        if (tid < off) sm[tid] += sm[tid + off];
        __syncthreads();
    }
    if (tid == 0) out[0] = sm[0] * invMA;
}

// ---------------------------------------------------------------------------
extern "C" void kernel_launch(void* const* d_in, const int* in_sizes, int n_in,
                              void* d_out, int out_size, void* d_ws, size_t ws_size,
                              hipStream_t stream)
{
    const float* emb    = (const float*)d_in[0];
    // d_in[1] (time_indices) is unused by the reference computation.
    const int*   negIdx = (const int*)d_in[2];

    const int M        = in_sizes[0] / DIM;          // 262144
    const int MA       = in_sizes[2] / NEGS;         // M - 1 anchors
    const int numTiles = (MA + TILE - 1) / TILE;

    const size_t bytesF16Path = (size_t)M * DIM * sizeof(_Float16)
                              + (size_t)numTiles * sizeof(float);

    if (ws_size >= bytesF16Path) {
        // Path A: normalized f16 mirror (64 MB, L2-resident) -> zero-math WMMA loads.
        _Float16* embh     = (_Float16*)d_ws;
        float*    partials = (float*)((char*)d_ws + (size_t)M * DIM * sizeof(_Float16));

        tcl_norm_f16_kernel<<<(M + 7) / 8, 256, 0, stream>>>(emb, embh, M);
        tcl_wmma_f16_kernel<<<(numTiles + 7) / 8, 256, 0, stream>>>(
            embh, negIdx, partials, M, MA, numTiles);
        tcl_reduce_kernel<<<1, 256, 0, stream>>>(partials, numTiles, (float*)d_out,
                                                 1.0f / (float)MA);
    } else {
        // Path B: tiny workspace fallback (inv-norm table, convert on the fly).
        float* invNorm  = (float*)d_ws;              // M floats
        float* partials = invNorm + M;               // numTiles floats

        tcl_norm_kernel<<<(M + 7) / 8, 256, 0, stream>>>(emb, invNorm, M);
        tcl_wmma_f32_kernel<<<(numTiles + 7) / 8, 256, 0, stream>>>(
            emb, negIdx, invNorm, partials, M, MA, numTiles);
        tcl_reduce_kernel<<<1, 256, 0, stream>>>(partials, numTiles, (float*)d_out,
                                                 1.0f / (float)MA);
    }
}
